// ProbSparseAttention_20830591385930
// MI455X (gfx1250) — matrix operand order repaired and checked by reading, not compiled, not run on hardware
//
#include <hip/hip_runtime.h>

typedef __bf16 bf16;
typedef __attribute__((ext_vector_type(16))) __bf16 v16bf;
typedef __attribute__((ext_vector_type(8)))  float  v8f;
typedef __attribute__((ext_vector_type(2)))  float  v2f;

#define B_    8
#define L_    4096
#define DM_   512
#define DK_   512
#define DV_   512
#define UK_   45
#define UPAD  48
#define SCALE_ 0.04419417382415922f   // 1/sqrt(512)

// ---------------------------------------------------------------------------
// Projection GEMM: out[r,c] = X[r,:] @ W[:,c] + bias[c]
// X: [B*L, 512] f32, W: [512, 512] f32. Block = 256 thr (8 waves), 64x64 tile.
// bf16 WMMA, f32 accumulate. LDS tiles stored so every fragment is two
// contiguous, 16B-aligned ds_load_b128's (A: [row][k], B: transposed [col][k]).
// ---------------------------------------------------------------------------
template <bool WB, bool WF>
__global__ __launch_bounds__(256) void proj_kernel(
    const float* __restrict__ X, const float* __restrict__ W,
    const float* __restrict__ bias, bf16* __restrict__ outb,
    float* __restrict__ outf)
{
    __shared__ bf16 As[64][40];   // [row][k], 64x32 used; 80B row stride (16B mult)
    __shared__ bf16 Bs[64][40];   // [col][k] (transposed), 64x32 used
    const int t    = threadIdx.x;
    const int lane = t & 31;
    const int wave = t >> 5;
    const int rowbase = blockIdx.x * 64;
    const int colbase = blockIdx.y * 64;
    const int mt   = wave & 3;            // row sub-tile 0..3
    const int nt0  = (wave >> 2) * 2;     // col sub-tile pair base
    const int m    = lane & 15;
    const int half = lane >> 4;

    v8f z = {};
    v8f acc0 = z, acc1 = z;

    for (int k0 = 0; k0 < DM_; k0 += 32) {
        #pragma unroll
        for (int i = 0; i < 8; ++i) {                 // 64x32 A tile
            int e = t + i * 256;
            int r = e >> 5, kk = e & 31;
            As[r][kk] = (bf16)X[(size_t)(rowbase + r) * DM_ + (k0 + kk)];
        }
        #pragma unroll
        for (int i = 0; i < 8; ++i) {                 // 32x64 W tile -> transposed LDS
            int e = t + i * 256;
            int kk = e >> 6, c = e & 63;              // consecutive threads: consecutive c (coalesced)
            Bs[c][kk] = (bf16)W[(size_t)(k0 + kk) * DK_ + (colbase + c)];
        }
        if (k0 + 32 < DM_) {                          // prefetch next K tile
            __builtin_prefetch(X + (size_t)(rowbase + (t >> 5)) * DM_ + (k0 + 32) + (t & 31), 0, 1);
            __builtin_prefetch(W + (size_t)(k0 + 32 + (t >> 6)) * DK_ + colbase + (t & 63), 0, 1);
        }
        __syncthreads();

        v16bf a, b0, b1;
        #pragma unroll
        for (int i = 0; i < 16; ++i) {
            a[i] = As[mt * 16 + m][16 * half + i];
        }
        #pragma unroll
        for (int i = 0; i < 16; ++i) {                // contiguous: merges to ds_load_b128 x2
            b0[i] = Bs[nt0 * 16 + m][16 * half + i];
            b1[i] = Bs[nt0 * 16 + 16 + m][16 * half + i];
        }
        acc0 = __builtin_amdgcn_wmma_f32_16x16x32_bf16(false, a, false, b0, (short)0, acc0, false, false);
        acc1 = __builtin_amdgcn_wmma_f32_16x16x32_bf16(false, a, false, b1, (short)0, acc1, false, false);
        __syncthreads();
    }

    #pragma unroll
    for (int v = 0; v < 8; ++v) {
        int r  = rowbase + mt * 16 + v + 8 * half;
        int c0 = colbase + nt0 * 16 + m;
        int c1 = c0 + 16;
        float x0 = acc0[v] + bias[c0];
        float x1 = acc1[v] + bias[c1];
        size_t o0 = (size_t)r * DK_ + c0;
        size_t o1 = (size_t)r * DK_ + c1;
        if (WF) { outf[o0] = x0;        outf[o1] = x1; }
        if (WB) { outb[o0] = (bf16)x0;  outb[o1] = (bf16)x1; }
    }
}

// ---------------------------------------------------------------------------
// sbar = Qp @ Kp[kidx].T  -> meas = rowmax - rowmean over the 45 sampled cols.
// Block = 128 thr (4 waves), 64 rows x 48 (padded) cols.
// ---------------------------------------------------------------------------
__global__ __launch_bounds__(128) void sbar_meas_kernel(
    const bf16* __restrict__ Qp, const bf16* __restrict__ Kp,
    const int* __restrict__ kidx, float* __restrict__ meas)
{
    __shared__ int   idx_s[UPAD];
    __shared__ float sb[64][UPAD + 1];
    const int b = blockIdx.y;
    const int rowbase = blockIdx.x * 64;
    const int t = threadIdx.x;
    const int lane = t & 31, wave = t >> 5;
    const int m = lane & 15, half = lane >> 4;
    if (t < UPAD) idx_s[t] = (t < UK_) ? kidx[b * UK_ + t] : -1;
    __syncthreads();

    const bf16* Qb = Qp + (size_t)b * L_ * DK_;
    const bf16* Kb = Kp + (size_t)b * L_ * DK_;
    const bf16* qrow = Qb + (size_t)(rowbase + wave * 16 + m) * DK_;
    const bf16* kq[3];
    #pragma unroll
    for (int nt = 0; nt < 3; ++nt) {
        int key = idx_s[nt * 16 + m];
        kq[nt] = (key >= 0) ? (Kb + (size_t)key * DK_) : (const bf16*)nullptr;
    }

    v8f z = {};
    v8f acc[3] = {z, z, z};
    for (int k0 = 0; k0 < DK_; k0 += 32) {
        v16bf a;
        #pragma unroll
        for (int i = 0; i < 16; ++i) {
            a[i] = qrow[k0 + 16 * half + i];
        }
        #pragma unroll
        for (int nt = 0; nt < 3; ++nt) {
            v16bf bb = {};
            if (kq[nt]) {
                #pragma unroll
                for (int i = 0; i < 16; ++i) bb[i] = kq[nt][k0 + 16 * half + i];
            }
            acc[nt] = __builtin_amdgcn_wmma_f32_16x16x32_bf16(false, a, false, bb, (short)0, acc[nt], false, false);
        }
    }
    #pragma unroll
    for (int nt = 0; nt < 3; ++nt)
        #pragma unroll
        for (int v = 0; v < 8; ++v)
            sb[wave * 16 + v + 8 * half][nt * 16 + m] = acc[nt][v];
    __syncthreads();

    if (t < 64) {
        float mx = -__builtin_inff(), sum = 0.f;
        for (int j = 0; j < UK_; ++j) { float x = sb[t][j]; mx = fmaxf(mx, x); sum += x; }
        meas[(size_t)b * L_ + rowbase + t] = mx - sum * (1.0f / UK_);
    }
}

// ---------------------------------------------------------------------------
// Top-45 indices of meas per batch; iterative argmax (ties -> lowest index,
// matching lax.top_k ordering). One block per batch.
// ---------------------------------------------------------------------------
__global__ __launch_bounds__(256) void topk_kernel(
    const float* __restrict__ meas, int* __restrict__ topI)
{
    __shared__ float vals[L_];
    __shared__ float rv[256];
    __shared__ int   ri[256];
    const int b = blockIdx.x, t = threadIdx.x;
    for (int i = t; i < L_; i += 256) vals[i] = meas[(size_t)b * L_ + i];
    __syncthreads();
    for (int it = 0; it < UK_; ++it) {
        float bv = -__builtin_inff(); int bi = 0x7fffffff;
        for (int i = t; i < L_; i += 256) {
            float v = vals[i];
            if (v > bv || (v == bv && i < bi)) { bv = v; bi = i; }
        }
        rv[t] = bv; ri[t] = bi;
        __syncthreads();
        for (int s = 128; s > 0; s >>= 1) {
            if (t < s) {
                float v = rv[t + s]; int ix = ri[t + s];
                if (v > rv[t] || (v == rv[t] && ix < ri[t])) { rv[t] = v; ri[t] = ix; }
            }
            __syncthreads();
        }
        if (t == 0) { topI[b * UPAD + it] = ri[0]; vals[ri[0]] = -__builtin_inff(); }
        __syncthreads();
    }
    if (t == 0) { topI[b * UPAD + 45] = 0; topI[b * UPAD + 46] = 0; topI[b * UPAD + 47] = 0; }
}

// ---------------------------------------------------------------------------
// logits[u, j] = SCALE * Qp[I[u],:] . Kp[j,:]   (48 padded rows x 4096)
// Block = 128 thr (4 waves): 48 rows x 64 cols; wave = one 16-col tile.
// ---------------------------------------------------------------------------
__global__ __launch_bounds__(128) void logits_kernel(
    const bf16* __restrict__ Qp, const bf16* __restrict__ Kp,
    const int* __restrict__ topI, float* __restrict__ logits)
{
    __shared__ int I_s[UPAD];
    const int b = blockIdx.y;
    const int colbase = blockIdx.x * 64;
    const int t = threadIdx.x;
    const int lane = t & 31, w = t >> 5;
    const int m = lane & 15, half = lane >> 4;
    if (t < UPAD) I_s[t] = topI[b * UPAD + t];
    __syncthreads();

    const bf16* Qb = Qp + (size_t)b * L_ * DK_;
    const bf16* Kb = Kp + (size_t)b * L_ * DK_;
    const bf16* qrow[3];
    #pragma unroll
    for (int mt = 0; mt < 3; ++mt) qrow[mt] = Qb + (size_t)I_s[mt * 16 + m] * DK_;
    const bf16* krow = Kb + (size_t)(colbase + w * 16 + m) * DK_;

    v8f z = {};
    v8f acc[3] = {z, z, z};
    for (int k0 = 0; k0 < DK_; k0 += 32) {
        v16bf bb;
        #pragma unroll
        for (int i = 0; i < 16; ++i) bb[i] = krow[k0 + 16 * half + i];
        #pragma unroll
        for (int mt = 0; mt < 3; ++mt) {
            v16bf a;
            #pragma unroll
            for (int i = 0; i < 16; ++i) {
                a[i] = qrow[mt][k0 + 16 * half + i];
            }
            acc[mt] = __builtin_amdgcn_wmma_f32_16x16x32_bf16(false, a, false, bb, (short)0, acc[mt], false, false);
        }
    }
    float* Lg = logits + (size_t)b * UPAD * L_;
    #pragma unroll
    for (int mt = 0; mt < 3; ++mt)
        #pragma unroll
        for (int v = 0; v < 8; ++v) {
            int u = mt * 16 + v + 8 * half;
            Lg[(size_t)u * L_ + (colbase + w * 16 + m)] = acc[mt][v] * SCALE_;
        }
}

// ---------------------------------------------------------------------------
// In-place stable softmax over each 4096-wide logit row. Grid = (48, B).
// ---------------------------------------------------------------------------
__global__ __launch_bounds__(256) void softmax_kernel(float* __restrict__ logits)
{
    __shared__ float red[256];
    const int b = blockIdx.y, u = blockIdx.x, t = threadIdx.x;
    float* row = logits + ((size_t)b * UPAD + u) * L_;
    float mx = -__builtin_inff();
    for (int i = t; i < L_; i += 256) mx = fmaxf(mx, row[i]);
    red[t] = mx; __syncthreads();
    for (int s = 128; s > 0; s >>= 1) { if (t < s) red[t] = fmaxf(red[t], red[t + s]); __syncthreads(); }
    mx = red[0]; __syncthreads();
    float sum = 0.f;
    for (int i = t; i < L_; i += 256) { float e = __expf(row[i] - mx); row[i] = e; sum += e; }
    red[t] = sum; __syncthreads();
    for (int s = 128; s > 0; s >>= 1) { if (t < s) red[t] += red[t + s]; __syncthreads(); }
    float inv = 1.0f / red[0];
    for (int i = t; i < L_; i += 256) row[i] *= inv;
}

// ---------------------------------------------------------------------------
// Base fill: out[b,i,:] = mean over features of Vp[b,i,:]. One wave per row.
// ---------------------------------------------------------------------------
__global__ __launch_bounds__(256) void rowmean_kernel(
    const float* __restrict__ Vp, float* __restrict__ out)
{
    const int row  = blockIdx.x * 8 + (threadIdx.x >> 5);   // over B*L rows
    const int lane = threadIdx.x & 31;
    const float* v = Vp + (size_t)row * DV_;
    float s = 0.f;
    for (int i = lane; i < DV_; i += 32) s += v[i];
    for (int off = 16; off > 0; off >>= 1) s += __shfl_down(s, off, 32);
    s = __shfl(s, 0, 32);
    const float mean = s * (1.0f / DV_);
    float* o = out + (size_t)row * DV_;
    for (int i = lane; i < DV_; i += 32) o[i] = mean;
}

// ---------------------------------------------------------------------------
// s1 = attn @ Vp in full fp32 via v_wmma_f32_16x16x4_f32; scatter rows at I.
// Grid = (3 mtiles, DV/64 col blocks, B); block = 128 thr, wave = 16 cols.
// ---------------------------------------------------------------------------
__global__ __launch_bounds__(128) void s1_kernel(
    const float* __restrict__ attn, const float* __restrict__ Vp,
    const int* __restrict__ topI, float* __restrict__ out)
{
    __shared__ int I_s[UPAD];
    const int b  = blockIdx.z;
    const int mt = blockIdx.x;
    const int colbase = blockIdx.y * 64;
    const int t = threadIdx.x;
    const int lane = t & 31, w = t >> 5;
    const int n = lane & 15, half = lane >> 4;
    if (t < UPAD) I_s[t] = topI[b * UPAD + t];
    __syncthreads();

    const float* A  = attn + (size_t)b * UPAD * L_;
    const float* Vb = Vp + (size_t)b * L_ * DV_;
    const int col  = colbase + w * 16 + n;
    const int arow = mt * 16 + n;

    v8f acc = {};
    for (int k0 = 0; k0 < L_; k0 += 4) {
        const int kk = k0 + 2 * half;      // A/B f32 frag: element v -> K = 2*half + v
        v2f a, bb;
        a[0]  = A[(size_t)arow * L_ + kk];
        a[1]  = A[(size_t)arow * L_ + kk + 1];
        bb[0] = Vb[(size_t)kk * DV_ + col];
        bb[1] = Vb[(size_t)(kk + 1) * DV_ + col];
        acc = __builtin_amdgcn_wmma_f32_16x16x4_f32(false, a, false, bb, (short)0, acc, false, false);
    }
    #pragma unroll
    for (int v = 0; v < 8; ++v) {
        int u = mt * 16 + v + 8 * half;
        if (u < UK_) {
            int r = I_s[u];
            out[(size_t)b * L_ * DV_ + (size_t)r * DV_ + col] = acc[v];
        }
    }
}

// ---------------------------------------------------------------------------
extern "C" void kernel_launch(void* const* d_in, const int* in_sizes, int n_in,
                              void* d_out, int out_size, void* d_ws, size_t ws_size,
                              hipStream_t stream)
{
    (void)in_sizes; (void)n_in; (void)out_size; (void)ws_size;
    const float* Q  = (const float*)d_in[0];
    const float* K  = (const float*)d_in[1];
    const float* V  = (const float*)d_in[2];
    const float* WQ = (const float*)d_in[3];
    const float* bQ = (const float*)d_in[4];
    const float* WK = (const float*)d_in[5];
    const float* bK = (const float*)d_in[6];
    const float* WV = (const float*)d_in[7];
    const float* bV = (const float*)d_in[8];
    const int* kidx = (const int*)d_in[9];
    float* out = (float*)d_out;

    char* ws = (char*)d_ws;
    const size_t szPbf = (size_t)B_ * L_ * DK_ * sizeof(bf16);    // 33,554,432
    const size_t szVp  = (size_t)B_ * L_ * DV_ * sizeof(float);   // 67,108,864
    const size_t szLg  = (size_t)B_ * UPAD * L_ * sizeof(float);  //  6,291,456
    bf16*  Qp   = (bf16*)ws;
    bf16*  Kp   = (bf16*)(ws + szPbf);
    float* Vp   = (float*)(ws + 2 * szPbf);
    float* lg   = (float*)(ws + 2 * szPbf + szVp);
    float* meas = (float*)(ws + 2 * szPbf + szVp + szLg);
    int*   topI = (int*)(ws + 2 * szPbf + szVp + szLg + (size_t)B_ * L_ * sizeof(float));

    dim3 gp((B_ * L_) / 64, DK_ / 64);
    proj_kernel<true,  false><<<gp, 256, 0, stream>>>(Q, WQ, bQ, Qp, (float*)nullptr);
    proj_kernel<true,  false><<<gp, 256, 0, stream>>>(K, WK, bK, Kp, (float*)nullptr);
    proj_kernel<false, true ><<<gp, 256, 0, stream>>>(V, WV, bV, (bf16*)nullptr, Vp);
    sbar_meas_kernel<<<dim3(L_ / 64, B_), 128, 0, stream>>>(Qp, Kp, kidx, meas);
    topk_kernel<<<B_, 256, 0, stream>>>(meas, topI);
    logits_kernel<<<dim3(L_ / 64, B_), 128, 0, stream>>>(Qp, Kp, topI, lg);
    softmax_kernel<<<dim3(UPAD, B_), 256, 0, stream>>>(lg);
    rowmean_kernel<<<(B_ * L_) / 8, 256, 0, stream>>>(Vp, out);
    s1_kernel<<<dim3(3, DV_ / 64, B_), 128, 0, stream>>>(lg, Vp, topI, out);
}
